// LinearSelfAttention_51969104281853
// MI455X (gfx1250) — matrix operand order repaired and checked by reading, not compiled, not run on hardware
//
#include <hip/hip_runtime.h>
#include <hip/hip_bf16.h>
#include <cstdint>

typedef __attribute__((ext_vector_type(16))) __bf16 v16bf;
typedef __attribute__((ext_vector_type(8)))  __bf16 v8bf;
typedef __attribute__((ext_vector_type(8)))  float  v8f;
typedef __attribute__((ext_vector_type(4)))  float  v4f;

#define HID   2048
#define S_LEN 4096
#define BATCH 4
#define HEADS 16
#define HD    128
#define MROWS (BATCH * S_LEN)   // 16384

__device__ __forceinline__ v16bf cat8(v8bf lo, v8bf hi) {
  return __builtin_shufflevector(lo, hi, 0,1,2,3,4,5,6,7,8,9,10,11,12,13,14,15);
}

__device__ __forceinline__ float elu1(float x) {
  return x > 0.f ? x + 1.f : __expf(x);   // elu(x)+1
}

// ---- CDNA5 async copy: global -> LDS (ASYNCcnt-tracked), inline asm ----
__device__ __forceinline__ void async_b128(void* lds_ptr, const void* gptr) {
  uint32_t lds_off = (uint32_t)(uintptr_t)lds_ptr;     // addr[31:0] = LDS offset
  uint64_t gaddr   = (uint64_t)(uintptr_t)gptr;
  asm volatile("global_load_async_to_lds_b128 %0, %1, off"
               :: "v"(lds_off), "v"(gaddr) : "memory");
}
__device__ __forceinline__ void wait_async0() {
  asm volatile("s_wait_asynccnt 0x0" ::: "memory");
}

// ---------------- fp32 -> bf16, 8 elems/thread ----------------
__global__ void cvt_bf16_kernel(const float* __restrict__ src,
                                __bf16* __restrict__ dst, int n8) {
  int i = blockIdx.x * blockDim.x + threadIdx.x;
  if (i < n8) {
    v4f a = ((const v4f*)src)[2 * i];
    v4f b = ((const v4f*)src)[2 * i + 1];
    v8bf o;
#pragma unroll
    for (int t = 0; t < 4; ++t) { o[t] = (__bf16)a[t]; o[t + 4] = (__bf16)b[t]; }
    ((v8bf*)dst)[i] = o;
  }
}

// ---------------- projection GEMM: Y = X @ W^T + b ----------------
// X: [M,2048] bf16 rm.  W: [N=2048,2048] bf16 rm.
// Double-buffered LDS tiles filled by global_load_async_to_lds_b128.
// MODE 0: elu+1 -> bf16 [B,H,S,D].  MODE 1: bf16 [B,H,S,D].  MODE 2: fp32 [M,2048].
template <int MODE>
__global__ void proj_kernel(const __bf16* __restrict__ X,
                            const __bf16* __restrict__ W,
                            const float* __restrict__ bias,
                            void* __restrict__ dst) {
  __shared__ __bf16 xs[2][128][40];    // 80B rows: 20-dword stride, conflict-free
  __shared__ __bf16 wt[2][128][40];
  const int tid  = threadIdx.x;
  const int lane = tid & 31;
  const int wid  = tid >> 5;
  const int lm   = lane & 15;
  const int lh   = lane >> 4;
  const int wm = (wid & 3) * 32;    // local M offset of this wave (4 waves)
  const int wn = (wid >> 2) * 64;   // local N offset of this wave (2 waves)
  const int sr = tid >> 1;          // staging row 0..127
  const int sc = (tid & 1) * 16;    // staging col (elem) 0/16

  const __bf16* xrow = X + (size_t)(blockIdx.x * 128 + sr) * HID + sc;
  const __bf16* wrow = W + (size_t)(blockIdx.y * 128 + sr) * HID + sc;

  auto stage = [&](int buf, int k0) {
    async_b128(&xs[buf][sr][sc],     xrow + k0);
    async_b128(&xs[buf][sr][sc + 8], xrow + k0 + 8);
    async_b128(&wt[buf][sr][sc],     wrow + k0);
    async_b128(&wt[buf][sr][sc + 8], wrow + k0 + 8);
  };

  v8f acc[2][4] = {};

  stage(0, 0);
  wait_async0();
  __syncthreads();

  for (int k0 = 0; k0 < HID; k0 += 32) {
    const int buf = (k0 >> 5) & 1;
    if (k0 + 32 < HID) stage(buf ^ 1, k0 + 32);   // fill next while computing

    v16bf af[2];
#pragma unroll
    for (int i = 0; i < 2; ++i) {
      // A 16x32: lanes 0-15 K in {0..7,16..23}; lanes 16-31 K in {8..15,24..31}
      const __bf16* row = &xs[buf][wm + i * 16 + lm][0];
      af[i] = cat8(*(const v8bf*)(row + lh * 8), *(const v8bf*)(row + 16 + lh * 8));
    }
    v16bf bfm[4];
#pragma unroll
    for (int j = 0; j < 4; ++j) {
      // B[k][n] = W[n][k]: lanes 0-15 (N=lm) K in {0..7,16..23}; lanes 16-31 K in {8..15,24..31}
      const __bf16* row = &wt[buf][wn + j * 16 + lm][0];
      bfm[j] = cat8(*(const v8bf*)(row + lh * 8), *(const v8bf*)(row + 16 + lh * 8));
    }
#pragma unroll
    for (int i = 0; i < 2; ++i)
#pragma unroll
      for (int j = 0; j < 4; ++j)
        acc[i][j] = __builtin_amdgcn_wmma_f32_16x16x32_bf16(
            false, af[i], false, bfm[j], (short)0, acc[i][j], false, false);

    wait_async0();     // next buffer filled
    __syncthreads();   // everyone done reading current buffer
  }

  const int gm0 = blockIdx.x * 128 + wm;
  const int gn0 = blockIdx.y * 128 + wn;
#pragma unroll
  for (int i = 0; i < 2; ++i)
#pragma unroll
    for (int j = 0; j < 4; ++j)
#pragma unroll
      for (int e = 0; e < 8; ++e) {
        const int m = gm0 + i * 16 + e + 8 * lh;
        const int n = gn0 + j * 16 + lm;
        float val = acc[i][j][e] + bias[n];
        if (MODE == 2) {
          ((float*)dst)[(size_t)m * HID + n] = val;
        } else {
          const int b = m >> 12, s = m & (S_LEN - 1);
          const int h = n >> 7, d = n & (HD - 1);
          if (MODE == 0) val = elu1(val);
          ((__bf16*)dst)[(((size_t)(b * HEADS + h)) * S_LEN + s) * HD + d] = (__bf16)val;
        }
      }
}

// ---------------- ksum[bh][d] = sum_s kp[bh][s][d] ----------------
__global__ void ksum_kernel(const __bf16* __restrict__ kp, float* __restrict__ ksum) {
  __shared__ float part[2][128];
  const int bh   = blockIdx.x;
  const int d    = threadIdx.x & 127;
  const int half = threadIdx.x >> 7;
  const __bf16* p = kp + (size_t)bh * S_LEN * HD + (size_t)half * (S_LEN / 2) * HD + d;
  float s = 0.f;
  for (int ss = 0; ss < S_LEN / 2; ++ss) s += (float)p[(size_t)ss * HD];
  part[half][d] = s;
  __syncthreads();
  if (threadIdx.x < 128) ksum[bh * HD + d] = part[0][d] + part[1][d];
}

// ---------------- kv[bh] = kp^T @ v  (128x4096 @ 4096x128) ----------------
// LDS holds TRANSPOSED tiles [d][s] / [e][s]; fragments are ds_load_b128s.
__global__ void kv_kernel(const __bf16* __restrict__ kp,
                          const __bf16* __restrict__ v,
                          float* __restrict__ kv) {
  __shared__ __bf16 kp_t[128][40];
  __shared__ __bf16 v_t[128][40];
  const int bh   = blockIdx.x;
  const int tid  = threadIdx.x;
  const int lane = tid & 31, wid = tid >> 5;
  const int lm = lane & 15, lh = lane >> 4;
  const int wm0 = (wid & 3) * 32;   // d rows
  const int wn0 = (wid >> 2) * 64;  // e cols
  const size_t base = (size_t)bh * S_LEN * HD;
  const int r  = tid >> 3;          // staged seq row 0..31
  const int c0 = (tid & 7) * 16;    // staged feature col start

  v8f acc[2][4] = {};

  for (int ks = 0; ks < S_LEN; ks += 32) {
    __syncthreads();
    {
      const __bf16* pk = kp + base + (size_t)(ks + r) * HD + c0;
      const __bf16* pv = v  + base + (size_t)(ks + r) * HD + c0;
      v8bf ka = *(const v8bf*)pk, kb = *(const v8bf*)(pk + 8);
      v8bf va = *(const v8bf*)pv, vb = *(const v8bf*)(pv + 8);
#pragma unroll
      for (int t = 0; t < 8; ++t) {
        kp_t[c0 + t][r]     = ka[t];
        kp_t[c0 + 8 + t][r] = kb[t];
        v_t[c0 + t][r]      = va[t];
        v_t[c0 + 8 + t][r]  = vb[t];
      }
    }
    __syncthreads();

    v16bf af[2];
#pragma unroll
    for (int i = 0; i < 2; ++i) {
      const __bf16* row = &kp_t[wm0 + i * 16 + lm][0];
      af[i] = cat8(*(const v8bf*)(row + lh * 8), *(const v8bf*)(row + 16 + lh * 8));
    }
#pragma unroll
    for (int j = 0; j < 4; ++j) {
      const __bf16* row = &v_t[wn0 + j * 16 + lm][0];
      const v16bf bmat =
          cat8(*(const v8bf*)(row + lh * 8), *(const v8bf*)(row + 16 + lh * 8));
#pragma unroll
      for (int i = 0; i < 2; ++i)
        acc[i][j] = __builtin_amdgcn_wmma_f32_16x16x32_bf16(
            false, af[i], false, bmat, (short)0, acc[i][j], false, false);
    }
  }

  float* out = kv + (size_t)bh * HD * HD;
#pragma unroll
  for (int i = 0; i < 2; ++i)
#pragma unroll
    for (int j = 0; j < 4; ++j)
#pragma unroll
      for (int e = 0; e < 8; ++e) {
        const int m = wm0 + i * 16 + e + 8 * lh;
        const int n = wn0 + j * 16 + lm;
        out[(size_t)m * HD + n] = acc[i][j][e];
      }
}

// ---------------- attn: ctx = (qp @ kv) / (qp @ ksum + eps) ----------------
__global__ void attn_kernel(const __bf16* __restrict__ qp,
                            const float* __restrict__ kv,
                            const float* __restrict__ ksum,
                            __bf16* __restrict__ ctx) {
  __shared__ __bf16 kvt[128][136];   // transposed kv, 272B rows
  __shared__ float z_s[128];
  __shared__ float ksum_s[128];
  const int bh  = blockIdx.y;
  const int s0  = blockIdx.x * 128;
  const int tid = threadIdx.x;
  const int lane = tid & 31, wid = tid >> 5;
  const int lm = lane & 15, lh = lane >> 4;

  const float* kvp = kv + (size_t)bh * HD * HD;
  for (int idx = tid; idx < HD * HD; idx += 256) {
    const int d = idx >> 7, e = idx & (HD - 1);
    kvt[e][d] = (__bf16)kvp[idx];
  }
  if (tid < HD) ksum_s[tid] = ksum[bh * HD + tid];
  __syncthreads();

  const size_t qbase = ((size_t)bh * S_LEN + s0) * HD;
  if (tid < 128) {
    const __bf16* qrow = qp + qbase + (size_t)tid * HD;
    float z = 0.f;
#pragma unroll
    for (int c = 0; c < HD; c += 8) {
      const v8bf q8 = *(const v8bf*)(qrow + c);
#pragma unroll
      for (int t = 0; t < 8; ++t) z += (float)q8[t] * ksum_s[c + t];
    }
    z_s[tid] = z;
  }
  __syncthreads();

  const int wm0 = wid * 16;  // 8 waves x 16 rows
  v8f acc[8] = {};
#pragma unroll
  for (int kk = 0; kk < HD; kk += 32) {
    const __bf16* p = qp + qbase + (size_t)(wm0 + lm) * HD + kk + lh * 8;
    const v16bf a = cat8(*(const v8bf*)p, *(const v8bf*)(p + 16));
#pragma unroll
    for (int j = 0; j < 8; ++j) {
      const __bf16* row = &kvt[j * 16 + lm][0];
      const v16bf bmat = cat8(*(const v8bf*)(row + kk + lh * 8),
                              *(const v8bf*)(row + kk + 16 + lh * 8));
      acc[j] = __builtin_amdgcn_wmma_f32_16x16x32_bf16(
          false, a, false, bmat, (short)0, acc[j], false, false);
    }
  }

  const int b = bh >> 4, h = bh & 15;
#pragma unroll
  for (int j = 0; j < 8; ++j)
#pragma unroll
    for (int e = 0; e < 8; ++e) {
      const int m = wm0 + e + 8 * lh;
      const int n = j * 16 + lm;
      const float val = acc[j][e] / (z_s[m] + 1e-6f);
      ctx[((size_t)(b * S_LEN + (s0 + m))) * HID + h * HD + n] = (__bf16)val;
    }
}

// ---------------- launcher ----------------
extern "C" void kernel_launch(void* const* d_in, const int* in_sizes, int n_in,
                              void* d_out, int out_size, void* d_ws, size_t ws_size,
                              hipStream_t stream) {
  (void)in_sizes; (void)n_in; (void)out_size; (void)ws_size;
  const float* hs  = (const float*)d_in[0];
  const float* q_w = (const float*)d_in[1];
  const float* q_b = (const float*)d_in[2];
  const float* k_w = (const float*)d_in[3];
  const float* k_b = (const float*)d_in[4];
  const float* v_w = (const float*)d_in[5];
  const float* v_b = (const float*)d_in[6];
  const float* o_w = (const float*)d_in[7];
  const float* o_b = (const float*)d_in[8];

  char* ws = (char*)d_ws;
  size_t off = 0;
  auto alloc = [&](size_t bytes) -> char* {
    char* p = ws + off;
    off += (bytes + 255) & ~(size_t)255;
    return p;
  };
  const size_t XN = (size_t)MROWS * HID;
  const size_t WN = (size_t)HID * HID;
  __bf16* Xb  = (__bf16*)alloc(XN * 2);
  __bf16* WQ  = (__bf16*)alloc(WN * 2);
  __bf16* WK  = (__bf16*)alloc(WN * 2);
  __bf16* WV  = (__bf16*)alloc(WN * 2);
  __bf16* WO  = (__bf16*)alloc(WN * 2);
  __bf16* QP  = (__bf16*)alloc(XN * 2);    // [B,H,S,D]
  __bf16* KP  = (__bf16*)alloc(XN * 2);
  __bf16* VH  = (__bf16*)alloc(XN * 2);
  float*  KV  = (float*)alloc((size_t)BATCH * HEADS * HD * HD * 4);
  float*  KS  = (float*)alloc((size_t)BATCH * HEADS * HD * 4);
  __bf16* CTX = (__bf16*)alloc(XN * 2);

  // 1) fp32 -> bf16
  cvt_bf16_kernel<<<(XN / 8 + 255) / 256, 256, 0, stream>>>(hs, Xb, (int)(XN / 8));
  cvt_bf16_kernel<<<(WN / 8 + 255) / 256, 256, 0, stream>>>(q_w, WQ, (int)(WN / 8));
  cvt_bf16_kernel<<<(WN / 8 + 255) / 256, 256, 0, stream>>>(k_w, WK, (int)(WN / 8));
  cvt_bf16_kernel<<<(WN / 8 + 255) / 256, 256, 0, stream>>>(v_w, WV, (int)(WN / 8));
  cvt_bf16_kernel<<<(WN / 8 + 255) / 256, 256, 0, stream>>>(o_w, WO, (int)(WN / 8));

  // 2) projections (bias + feature map + head reshape fused)
  dim3 pgrid(MROWS / 128, HID / 128);
  proj_kernel<0><<<pgrid, 256, 0, stream>>>(Xb, WQ, q_b, (void*)QP);
  proj_kernel<0><<<pgrid, 256, 0, stream>>>(Xb, WK, k_b, (void*)KP);
  proj_kernel<1><<<pgrid, 256, 0, stream>>>(Xb, WV, v_b, (void*)VH);

  // 3) ksum, 4) kv
  ksum_kernel<<<BATCH * HEADS, 256, 0, stream>>>(KP, KS);
  kv_kernel<<<BATCH * HEADS, 256, 0, stream>>>(KP, VH, KV);

  // 5) attention application
  attn_kernel<<<dim3(S_LEN / 128, BATCH * HEADS), 256, 0, stream>>>(QP, KV, KS, CTX);

  // 6) output projection -> fp32 d_out
  proj_kernel<2><<<pgrid, 256, 0, stream>>>(CTX, WO, o_b, d_out);
}
